// TrendDiffSegARBlock_8014408974631
// MI455X (gfx1250) — matrix operand order repaired and checked by reading, not compiled, not run on hardware
//
#include <hip/hip_runtime.h>
#include <hip/hip_bf16.h>

typedef _Float16 f16;
typedef __attribute__((ext_vector_type(16))) _Float16 v16h;
typedef __attribute__((ext_vector_type(8)))  _Float16 v8h;
typedef __attribute__((ext_vector_type(8)))  float    v8f;
typedef __attribute__((ext_vector_type(4)))  int      v4i;

#define NN      1024   // nodes
#define NB      8      // batch
#define FREQ    42     // segments
#define TREC    41     // stconv T for reconstruction
#define TPADREC 51     // 41 + 2*5 halo
#define SEQT    28     // AR sequence buffer: 11 window + 12 preds + 5 halo
#define STEPS   12
#define MAXKT   11
#define DC      64     // d_conv
#define DM      128    // d_model
#define KINC    704    // 11 taps * 64
#define KPROJ   320    // 5 blocks * 64 (tc shared between both mixprops)
#define ALPHA_F 0.05f

#define KSLAB   128    // K-slab staged into LDS in k_hop
#define LDSROW  136    // padded LDS row stride (halves): 68 dwords -> conflict-free

#if defined(__gfx1250__) && __has_builtin(__builtin_amdgcn_global_load_async_to_lds_b128) && __has_builtin(__builtin_amdgcn_s_wait_asynccnt)
#define ASYNC_LDS 1
#endif

// ---------- WMMA helpers (CDNA5 16x16x32 f16 -> f32) ----------
// A frag (16x32, ISA 7.12.2): lane 0-15 M=lane holds K {0..7,16..23};
// lanes 16-31 hold K {8..15,24..31}  -> two 16B loads at +0 and +16 halves.
static __device__ __forceinline__ v16h ld_a(const f16* p) {
  union { v16h v; v8h h[2]; } u;
  u.h[0] = *(const v8h*)(p);
  u.h[1] = *(const v8h*)(p + 16);
  return u.v;
}
// B frag (32x16): lane 0-15 N=lane holds K {0..7,16..23}; lanes 16-31 hold K {8..15,24..31}
// -> same K order as A, two 16B loads at +0 and +16 halves.
static __device__ __forceinline__ v16h ld_b(const f16* p) {
  union { v16h v; v8h h[2]; } u;
  u.h[0] = *(const v8h*)(p);
  u.h[1] = *(const v8h*)(p + 16);
  return u.v;
}
#define WMMA(a, b, c) __builtin_amdgcn_wmma_f32_16x16x32_f16(false, (a), false, (b), (short)0, (c), false, false)

#define WAVE_SETUP() \
  const int lane = threadIdx.x & 31;            \
  const int lr   = lane & 15;                   \
  const int asub = (lane & 16) ? 8  : 0;        \
  const int bsub = (lane & 16) ? 8  : 0;        \
  const int w    = blockIdx.x * 8 + (threadIdx.x >> 5);
// C/D layout: lane N = lr; VGPR r holds M = asub + r (asub doubles as C-row offset).

#ifdef ASYNC_LDS
// builtin signature (from clang-22 diagnostic): arg0 = int4* in global AS(1) (non-const),
// arg1 = int4* in shared AS(3), then imm offset, imm cpol.
static __device__ __forceinline__ void cp_async_b128(const f16* g, f16* l) {
  __builtin_amdgcn_global_load_async_to_lds_b128(
      (__attribute__((address_space(1))) v4i*)g,
      (__attribute__((address_space(3))) v4i*)l, 0, 0);
}
#endif

// ---------- adjacency normalization ----------
__global__ void __launch_bounds__(256) k_adj_sums(const float* __restrict__ adj,
                                                  float* rowP, float* colP) {
  __shared__ float sr[256], sc[256];
  const int v = blockIdx.x, tid = threadIdx.x;
  float rs = 0.f, cs = 0.f;
  for (int n = tid; n < NN; n += 256) {
    rs += adj[(size_t)v * NN + n];
    cs += adj[(size_t)n * NN + v];
  }
  sr[tid] = rs; sc[tid] = cs;
  __syncthreads();
  for (int s = 128; s > 0; s >>= 1) {
    if (tid < s) { sr[tid] += sr[tid + s]; sc[tid] += sc[tid + s]; }
    __syncthreads();
  }
  if (tid == 0) { rowP[v] = sr[0] + 1.0f; colP[v] = sc[0] + 1.0f; }
}

// a1s = 0.95 * norm_adj(adj), a2s = 0.95 * norm_adj(adj^T)   (f16, (1-alpha) folded in)
__global__ void __launch_bounds__(256) k_adj_fill(const float* __restrict__ adj,
                                                  const float* __restrict__ rowP,
                                                  const float* __restrict__ colP,
                                                  f16* a1s, f16* a2s) {
  const int idx = blockIdx.x * 256 + threadIdx.x;
  const int v = idx >> 10, n = idx & 1023;
  const float diag = (v == n) ? 1.0f : 0.0f;
  a1s[idx] = (f16)(0.95f * (adj[(size_t)v * NN + n] + diag) / rowP[v]);
  a2s[idx] = (f16)(0.95f * (adj[(size_t)n * NN + v] + diag) / colP[v]);
}

// ---------- weight preparation (transposed f16 B-operands) ----------
struct PrepArgs {
  const float* g1w; const float* g1b; const float* g2w; const float* g2b;
  const float* encw2; const float* decw1; const float* decw2;
  const float* cw[6]; const float* cb[6];
};
__global__ void __launch_bounds__(256) k_prep(PrepArgs pa, f16* WincT, float* biasInc,
                                              f16* WprojT, float* biasProj,
                                              f16* encW2T, f16* decW1T, f16* decW2T) {
  const int gid = blockIdx.x * blockDim.x + threadIdx.x;
  const int stride = gridDim.x * blockDim.x;
  // Merged temporal-inception taps: WincT[o][didx*64+ci] = (1/6) sum_i w_i[o][ci][didx-5+i]
  for (int idx = gid; idx < 64 * KINC; idx += stride) {
    const int o = idx / KINC, kk = idx % KINC;
    const int didx = kk >> 6, ci = kk & 63;
    const int delta = didx - 5;
    const int ad = delta < 0 ? -delta : delta;
    float s = 0.f;
    for (int i = 0; i < 6; ++i)
      if (ad <= i) s += pa.cw[i][(size_t)(o * 64 + ci) * (2 * i + 1) + (delta + i)];
    WincT[idx] = (f16)(s * (1.0f / 6.0f));
  }
  for (int o = gid; o < 64; o += stride) {
    float s = 0.f;
    for (int i = 0; i < 6; ++i) s += pa.cb[i][o];
    biasInc[o]  = s * (1.0f / 6.0f);
    biasProj[o] = pa.g1b[o] + pa.g2b[o];
  }
  // Fused mixprop projection: blocks [tc(sum of both W0), h1a, h2a, h1b, h2b]
  for (int idx = gid; idx < 64 * KPROJ; idx += stride) {
    const int o = idx / KPROJ, kk = idx % KPROJ;
    const int blk = kk >> 6, ci = kk & 63;
    float v;
    if      (blk == 0) v = pa.g1w[ci * 64 + o] + pa.g2w[ci * 64 + o];
    else if (blk == 1) v = pa.g1w[(64  + ci) * 64 + o];
    else if (blk == 2) v = pa.g1w[(128 + ci) * 64 + o];
    else if (blk == 3) v = pa.g2w[(64  + ci) * 64 + o];
    else               v = pa.g2w[(128 + ci) * 64 + o];
    WprojT[idx] = (f16)v;
  }
  for (int idx = gid; idx < 64 * 128; idx += stride) {          // encW2T (64 x 128)
    const int o = idx / 128, k = idx & 127;
    encW2T[idx] = (f16)pa.encw2[k * 64 + o];
  }
  for (int idx = gid; idx < 128 * 64; idx += stride) {          // decW1T (128 x 64)
    const int o = idx / 64, k = idx & 63;
    decW1T[idx] = (f16)pa.decw1[k * 128 + o];
  }
  for (int idx = gid; idx < 16 * 128; idx += stride) {          // decW2T (16 x 128, rows>=8 zero)
    const int o = idx / 128, k = idx & 127;
    decW2T[idx] = (o < 8) ? (f16)pa.decw2[k * 8 + o] : (f16)0.0f;
  }
}

// ---------- encoder layer 1 (K=8, VALU; trivial fraction of FLOPs) ----------
__global__ void __launch_bounds__(256) k_enc1(const float* __restrict__ x_diff,
                                              const float* __restrict__ w1,
                                              const float* __restrict__ b1,
                                              f16* __restrict__ hidden) {
  __shared__ float xs[512], ws[1024], bs[128];
  const int tid = threadIdx.x;
  const int bt = blockIdx.x >> 4, vc = blockIdx.x & 15;
  const int b = bt / FREQ, f = bt % FREQ;
  for (int i = tid; i < 512; i += 256) {
    const int s = i >> 6, vl = i & 63;
    xs[i] = x_diff[((size_t)(b * 336 + f * 8 + s)) * NN + vc * 64 + vl];
  }
  for (int i = tid; i < 1024; i += 256) ws[i] = w1[i];
  if (tid < 128) bs[tid] = b1[tid];
  __syncthreads();
  for (int i = tid; i < 64 * 128; i += 256) {
    const int vl = i >> 7, j = i & 127;
    float s = bs[j];
    #pragma unroll
    for (int t = 0; t < 8; ++t) s += xs[t * 64 + vl] * ws[t * 128 + j];
    hidden[((size_t)bt * NN + vc * 64 + vl) * DM + j] = (f16)fmaxf(s, 0.0f);
  }
}

// ---------- encoder layer 2 (WMMA, K=128) + fanout of res into all consumers ----------
__global__ void __launch_bounds__(256) k_enc2(const f16* __restrict__ hidden,
                                              const f16* __restrict__ encW2T,
                                              const float* __restrict__ b2,
                                              float* __restrict__ resF32,
                                              f16* __restrict__ X16rec,
                                              f16* __restrict__ seq16,
                                              f16* __restrict__ rec16) {
  WAVE_SETUP();
  const int Rbase = w * 16;
  const f16* Arow = hidden + (size_t)(Rbase + lr) * DM;
  v8f acc[4] = {};
  #pragma unroll
  for (int k0 = 0; k0 < DM; k0 += 32) {
    const v16h a = ld_a(Arow + k0 + asub);
    #pragma unroll
    for (int nt = 0; nt < 4; ++nt) {
      const v16h bb = ld_b(encW2T + (size_t)(nt * 16 + lr) * DM + k0 + bsub);
      acc[nt] = WMMA(a, bb, acc[nt]);
    }
  }
  const int bt = Rbase >> 10, b = bt / FREQ, f = bt % FREQ;
  #pragma unroll
  for (int nt = 0; nt < 4; ++nt) {
    const int d = nt * 16 + lr;
    #pragma unroll
    for (int r = 0; r < 8; ++r) {
      const int R = Rbase + asub + r, v = R & 1023;
      const float val = acc[nt][r] + b2[d];
      resF32[(size_t)R * DC + d] = val;
      const f16 hv = (f16)val;
      if (f < TREC)
        X16rec[(((size_t)(b * TPADREC + f + 5)) * NN + v) * DC + d] = hv;
      if (f >= FREQ - MAXKT)
        seq16[(((size_t)(b * SEQT + (f - (FREQ - MAXKT)))) * NN + v) * DC + d] = hv;
      if (f == 0)
        rec16[(((size_t)(b * FREQ)) * NN + v) * DC + d] = hv;
    }
  }
}

// ---------- temporal inception as one K=704 WMMA GEMM per (b,t) ----------
__global__ void __launch_bounds__(256) k_inc(const f16* __restrict__ X, int Tpad, int tdiv,
                                             int toff, int startPair,
                                             const f16* __restrict__ WincT,
                                             const float* __restrict__ biasInc,
                                             f16* __restrict__ tcrow, f16* __restrict__ tcT) {
  WAVE_SETUP();
  const int pair = w >> 6, mtile = w & 63;
  const int vbase = mtile * 16, v = vbase + lr;
  const int pg = startPair + pair;
  int b, t0;
  if (tdiv > 0) { b = pg / tdiv; t0 = pg % tdiv; } else { b = pg; t0 = toff; }
  v8f acc[4] = {};
  #pragma unroll 2
  for (int k0 = 0; k0 < KINC; k0 += 32) {
    const int didx = k0 >> 6, ci = k0 & 63;
    const f16* ap = X + (((size_t)(b * Tpad + t0 + didx)) * NN + v) * DC + ci + asub;
    const v16h a = ld_a(ap);
    #pragma unroll
    for (int nt = 0; nt < 4; ++nt) {
      const v16h bb = ld_b(WincT + (size_t)(nt * 16 + lr) * KINC + k0 + bsub);
      acc[nt] = WMMA(a, bb, acc[nt]);
    }
  }
  f16* tr = tcrow + (size_t)pair * (NN * DC);
  f16* tt = tcT   + (size_t)pair * (DC * NN);
  #pragma unroll
  for (int nt = 0; nt < 4; ++nt) {
    const int d = nt * 16 + lr;
    v8h pk;
    #pragma unroll
    for (int r = 0; r < 8; ++r) {
      const int vv = vbase + asub + r;
      const float h = acc[nt][r] + biasInc[d];
      tr[(size_t)vv * DC + d] = (f16)h;
      pk[r] = (f16)h;
    }
    *(v8h*)(tt + (size_t)d * NN + vbase + asub) = pk;   // 16B-aligned packed store
  }
}

// ---------- graph hop: H_out = alpha*tc + (A_scaled @ H_in)   (WMMA, K=1024) ----------
// All 8 waves of a block share one (b,t) pair => B (H^T, 128KB) is identical for the
// whole block: stage K-slabs (64 x 128 f16, padded) into LDS, double-buffered via the
// CDNA5 async-to-LDS DMA path (ASYNCcnt) so the copy of slab s+1 overlaps WMMAs of slab s.
__global__ void __launch_bounds__(256) k_hop(const f16* __restrict__ Amat,
                                             const f16* __restrict__ HTin,
                                             const f16* __restrict__ Xrow,
                                             f16* __restrict__ Hrow,
                                             f16* __restrict__ HTout) {
  __shared__ f16 Bs[2 * 64 * LDSROW];
  WAVE_SETUP();
  const int tid = threadIdx.x;
  const int pair = w >> 6, mtile = w & 63;
  const int vbase = mtile * 16;
  const f16* Arow  = Amat + (size_t)(vbase + lr) * NN;
  const f16* Bbase = HTin + (size_t)pair * (DC * NN);
  v8f acc[4] = {};

#ifdef ASYNC_LDS
  // prologue: slab 0 -> buffer 0 (4 async b128 per wave-lane-set)
  #pragma unroll
  for (int j = 0; j < 4; ++j) {
    const int idx = j * 256 + tid, d = idx >> 4, c = idx & 15;
    cp_async_b128(Bbase + (size_t)d * NN + c * 8, &Bs[d * LDSROW + c * 8]);
  }
  for (int s = 0; s < NN / KSLAB; ++s) {
    const int buf = (s & 1) * (64 * LDSROW);
    if (s + 1 < NN / KSLAB) {
      const int nbuf = ((s + 1) & 1) * (64 * LDSROW);
      #pragma unroll
      for (int j = 0; j < 4; ++j) {
        const int idx = j * 256 + tid, d = idx >> 4, c = idx & 15;
        cp_async_b128(Bbase + (size_t)d * NN + (s + 1) * KSLAB + c * 8,
                      &Bs[nbuf + d * LDSROW + c * 8]);
      }
      __builtin_amdgcn_s_wait_asynccnt(4);   // slab s landed; slab s+1 in flight
    } else {
      __builtin_amdgcn_s_wait_asynccnt(0);
    }
    __syncthreads();
    #pragma unroll
    for (int kk = 0; kk < KSLAB; kk += 32) {
      const v16h a = ld_a(Arow + s * KSLAB + kk + asub);
      #pragma unroll
      for (int nt = 0; nt < 4; ++nt) {
        const v16h bb = ld_b(&Bs[buf + (nt * 16 + lr) * LDSROW + kk + bsub]);
        acc[nt] = WMMA(a, bb, acc[nt]);
      }
    }
    __syncthreads();
  }
#else
  // fallback: stage through VGPRs into LDS (still 8x B-traffic reduction)
  for (int s = 0; s < NN / KSLAB; ++s) {
    #pragma unroll
    for (int j = 0; j < 4; ++j) {
      const int idx = j * 256 + tid, d = idx >> 4, c = idx & 15;
      *(v8h*)&Bs[d * LDSROW + c * 8] =
          *(const v8h*)(Bbase + (size_t)d * NN + s * KSLAB + c * 8);
    }
    __syncthreads();
    #pragma unroll
    for (int kk = 0; kk < KSLAB; kk += 32) {
      const v16h a = ld_a(Arow + s * KSLAB + kk + asub);
      #pragma unroll
      for (int nt = 0; nt < 4; ++nt) {
        const v16h bb = ld_b(&Bs[(nt * 16 + lr) * LDSROW + kk + bsub]);
        acc[nt] = WMMA(a, bb, acc[nt]);
      }
    }
    __syncthreads();
  }
#endif

  const f16* Xr = Xrow + (size_t)pair * (NN * DC);
  f16* Hr = Hrow + (size_t)pair * (NN * DC);
  #pragma unroll
  for (int nt = 0; nt < 4; ++nt) {
    const int d = nt * 16 + lr;
    v8h pk;
    #pragma unroll
    for (int r = 0; r < 8; ++r) {
      const int vv = vbase + asub + r;
      const float h = ALPHA_F * (float)Xr[(size_t)vv * DC + d] + acc[nt][r];
      Hr[(size_t)vv * DC + d] = (f16)h;
      pk[r] = (f16)h;
    }
    if (HTout)
      *(v8h*)(HTout + (size_t)pair * (DC * NN) + (size_t)d * NN + vbase + asub) = pk;
  }
}

// ---------- fused projection of both mixprops + biases + residual (WMMA, K=320) ----------
__global__ void __launch_bounds__(256) k_proj(const f16* __restrict__ tcrow,
                                              const f16* __restrict__ h1a,
                                              const f16* __restrict__ h2a,
                                              const f16* __restrict__ h1b,
                                              const f16* __restrict__ h2b,
                                              const f16* __restrict__ WprojT,
                                              const float* __restrict__ biasProj,
                                              int mode, int step, int startPair,
                                              const float* __restrict__ resF32,
                                              f16* __restrict__ seq16,
                                              f16* __restrict__ rec16) {
  WAVE_SETUP();
  const int pair = w >> 6, mtile = w & 63;
  const int vbase = mtile * 16;
  const int pg = startPair + pair;
  int b, t0;
  if (mode == 0) { b = pg / TREC; t0 = pg % TREC; } else { b = pg; t0 = 0; }
  const f16* Ab[5] = { tcrow, h1a, h2a, h1b, h2b };
  v8f acc[4] = {};
  #pragma unroll
  for (int k0 = 0; k0 < KPROJ; k0 += 32) {
    const int blk = k0 >> 6, cc = k0 & 63;
    const f16* ap = Ab[blk] + (size_t)pair * (NN * DC) + (size_t)(vbase + lr) * DC + cc + asub;
    const v16h a = ld_a(ap);
    #pragma unroll
    for (int nt = 0; nt < 4; ++nt) {
      const v16h bb = ld_b(WprojT + (size_t)(nt * 16 + lr) * KPROJ + k0 + bsub);
      acc[nt] = WMMA(a, bb, acc[nt]);
    }
  }
  #pragma unroll
  for (int nt = 0; nt < 4; ++nt) {
    const int d = nt * 16 + lr;
    #pragma unroll
    for (int r = 0; r < 8; ++r) {
      const int vv = vbase + asub + r;
      float resid;
      if (mode == 0)
        resid = resF32[(((size_t)(b * FREQ + t0)) * NN + vv) * DC + d];
      else
        resid = (float)seq16[(((size_t)(b * SEQT + step + 10)) * NN + vv) * DC + d];
      const float val = acc[nt][r] + biasProj[d] + resid;
      if (mode == 0)
        rec16[(((size_t)(b * FREQ + t0 + 1)) * NN + vv) * DC + d] = (f16)val;
      else
        seq16[(((size_t)(b * SEQT + MAXKT + step)) * NN + vv) * DC + d] = (f16)val;
    }
  }
}

// ---------- decoder layer 1 (WMMA, K=64 -> 128, ReLU) ----------
__global__ void __launch_bounds__(256) k_dec1(const f16* __restrict__ In,
                                              const f16* __restrict__ decW1T,
                                              const float* __restrict__ b1,
                                              f16* __restrict__ hidden,
                                              int TBT, int TSTR, int TOFF) {
  WAVE_SETUP();
  const int mtile = w >> 1, ng = w & 1;
  const int Rbase = mtile * 16;
  const int bt = Rbase >> 10, b = bt / TBT, t = bt % TBT;
  const f16* Abase = In + (((size_t)(b * TSTR + TOFF + t)) * NN) * DC;
  const f16* Arow  = Abase + (size_t)((Rbase + lr) & 1023) * DC;
  v8f acc[4] = {};
  #pragma unroll
  for (int k0 = 0; k0 < DC; k0 += 32) {
    const v16h a = ld_a(Arow + k0 + asub);
    #pragma unroll
    for (int nt = 0; nt < 4; ++nt) {
      const v16h bb = ld_b(decW1T + (size_t)(ng * 64 + nt * 16 + lr) * DC + k0 + bsub);
      acc[nt] = WMMA(a, bb, acc[nt]);
    }
  }
  #pragma unroll
  for (int nt = 0; nt < 4; ++nt) {
    const int o = ng * 64 + nt * 16 + lr;
    #pragma unroll
    for (int r = 0; r < 8; ++r) {
      const float val = fmaxf(acc[nt][r] + b1[o], 0.0f);
      hidden[(size_t)(Rbase + asub + r) * DM + o] = (f16)val;
    }
  }
}

// ---------- decoder layer 2 (WMMA, K=128 -> 8, scatter to time-major output) ----------
__global__ void __launch_bounds__(256) k_dec2(const f16* __restrict__ hidden,
                                              const f16* __restrict__ decW2T,
                                              const float* __restrict__ b2,
                                              float* __restrict__ out,
                                              int TBT, int OUTT, unsigned long long outBase) {
  WAVE_SETUP();
  const int Rbase = w * 16;
  const int bt = Rbase >> 10, b = bt / TBT, f = bt % TBT;
  const f16* Arow = hidden + (size_t)(Rbase + lr) * DM;
  v8f acc = {};
  #pragma unroll
  for (int k0 = 0; k0 < DM; k0 += 32) {
    const v16h a = ld_a(Arow + k0 + asub);
    const v16h bb = ld_b(decW2T + (size_t)lr * DM + k0 + bsub);
    acc = WMMA(a, bb, acc);
  }
  const int s = lr;
  if (s < 8) {
    #pragma unroll
    for (int r = 0; r < 8; ++r) {
      const int v = (Rbase + asub + r) & 1023;
      out[(size_t)outBase + ((size_t)(b * OUTT + f * 8 + s)) * NN + v] = acc[r] + b2[s];
    }
  }
}

// =====================================================================
extern "C" void kernel_launch(void* const* d_in, const int* in_sizes, int n_in,
                              void* d_out, int out_size, void* d_ws, size_t ws_size,
                              hipStream_t stream) {
  const float* x_diff = (const float*)d_in[0];
  const float* adj    = (const float*)d_in[1];
  const float* enc_w1 = (const float*)d_in[2];
  const float* enc_b1 = (const float*)d_in[3];
  const float* enc_w2 = (const float*)d_in[4];
  const float* enc_b2 = (const float*)d_in[5];
  const float* dec_w1 = (const float*)d_in[6];
  const float* dec_b1 = (const float*)d_in[7];
  const float* dec_w2 = (const float*)d_in[8];
  const float* dec_b2 = (const float*)d_in[9];
  PrepArgs pa;
  pa.g1w = (const float*)d_in[10]; pa.g1b = (const float*)d_in[11];
  pa.g2w = (const float*)d_in[12]; pa.g2b = (const float*)d_in[13];
  pa.encw2 = enc_w2; pa.decw1 = dec_w1; pa.decw2 = dec_w2;
  for (int i = 0; i < 6; ++i) {
    pa.cw[i] = (const float*)d_in[14 + 2 * i];
    pa.cb[i] = (const float*)d_in[15 + 2 * i];
  }

  // -------- workspace bump allocator --------
  size_t off = 0;
  auto alloc = [&](size_t bytes) -> void* {
    off = (off + 255) & ~(size_t)255;
    void* p = (char*)d_ws + off;
    off += bytes;
    return p;
  };
  const size_t ROWS  = (size_t)NB * FREQ * NN;          // 344064 encoder/decoder rows
  const size_t PROWS = (size_t)NB * STEPS * NN;         // 98304 prediction rows
  const int    CH    = 64;                              // (b,t) pairs per stconv chunk

  f16*   a1s     = (f16*)alloc((size_t)NN * NN * 2);
  f16*   a2s     = (f16*)alloc((size_t)NN * NN * 2);
  f16*   WincT   = (f16*)alloc((size_t)64 * KINC * 2);
  f16*   WprojT  = (f16*)alloc((size_t)64 * KPROJ * 2);
  f16*   encW2T  = (f16*)alloc((size_t)64 * 128 * 2);
  f16*   decW1T  = (f16*)alloc((size_t)128 * 64 * 2);
  f16*   decW2T  = (f16*)alloc((size_t)16 * 128 * 2);
  float* biasInc = (float*)alloc(64 * 4);
  float* biasProj= (float*)alloc(64 * 4);
  float* rowP    = (float*)alloc(NN * 4);
  float* colP    = (float*)alloc(NN * 4);
  f16*   hidden16= (f16*)alloc(ROWS * DM * 2);          // reused by decoder
  float* resF32  = (float*)alloc(ROWS * DC * 4);
  f16*   X16rec  = (f16*)alloc((size_t)NB * TPADREC * NN * DC * 2);
  f16*   seq16   = (f16*)alloc((size_t)NB * SEQT * NN * DC * 2);
  f16*   rec16   = (f16*)alloc((size_t)NB * FREQ * NN * DC * 2);
  const size_t SB = (size_t)CH * NN * DC * 2;           // per-buffer scratch bytes
  f16* tcrow  = (f16*)alloc(SB);
  f16* tcT    = (f16*)alloc(SB);
  f16* h1aT   = (f16*)alloc(SB);
  f16* h1arow = (f16*)alloc(SB);
  f16* h1bT   = (f16*)alloc(SB);
  f16* h1brow = (f16*)alloc(SB);
  f16* h2arow = (f16*)alloc(SB);
  f16* h2brow = (f16*)alloc(SB);
  (void)ws_size; (void)in_sizes; (void)n_in; (void)out_size;

  // zero-padded staging buffers (halo slots / not-yet-written AR slots must read as 0)
  (void)hipMemsetAsync(X16rec, 0, (size_t)NB * TPADREC * NN * DC * 2, stream);
  (void)hipMemsetAsync(seq16,  0, (size_t)NB * SEQT * NN * DC * 2,  stream);

  // -------- preprocessing --------
  k_adj_sums<<<NN, 256, 0, stream>>>(adj, rowP, colP);
  k_adj_fill<<<(NN * NN) / 256, 256, 0, stream>>>(adj, rowP, colP, a1s, a2s);
  k_prep<<<64, 256, 0, stream>>>(pa, WincT, biasInc, WprojT, biasProj, encW2T, decW1T, decW2T);

  // -------- encoder --------
  k_enc1<<<NB * FREQ * 16, 256, 0, stream>>>(x_diff, enc_w1, enc_b1, hidden16);
  k_enc2<<<(unsigned)(ROWS / 16 / 8), 256, 0, stream>>>(hidden16, encW2T, enc_b2,
                                                        resF32, X16rec, seq16, rec16);

  // -------- reconstruction stconv (chunked over (b,t) pairs) --------
  const int totalPairs = NB * TREC;                     // 328
  for (int st = 0; st < totalPairs; st += CH) {
    const int pairs = (totalPairs - st < CH) ? (totalPairs - st) : CH;
    const int blocks = pairs * 8;                       // pairs*64 mtiles / 8 waves
    k_inc <<<blocks, 256, 0, stream>>>(X16rec, TPADREC, TREC, 0, st, WincT, biasInc, tcrow, tcT);
    k_hop <<<blocks, 256, 0, stream>>>(a1s, tcT,  tcrow, h1arow, h1aT);
    k_hop <<<blocks, 256, 0, stream>>>(a2s, tcT,  tcrow, h1brow, h1bT);
    k_hop <<<blocks, 256, 0, stream>>>(a1s, h1aT, tcrow, h2arow, (f16*)nullptr);
    k_hop <<<blocks, 256, 0, stream>>>(a2s, h1bT, tcrow, h2brow, (f16*)nullptr);
    k_proj<<<blocks, 256, 0, stream>>>(tcrow, h1arow, h2arow, h1brow, h2brow, WprojT, biasProj,
                                       0, 0, st, resF32, seq16, rec16);
  }

  // -------- decode reconstruction --------
  k_dec1<<<(unsigned)((ROWS / 16) * 2 / 8), 256, 0, stream>>>(rec16, decW1T, dec_b1, hidden16,
                                                              FREQ, FREQ, 0);
  k_dec2<<<(unsigned)(ROWS / 16 / 8), 256, 0, stream>>>(hidden16, decW2T, dec_b2, (float*)d_out,
                                                        FREQ, 336, 0ULL);

  // -------- autoregressive prediction (only the consumed last timestep computed) --------
  for (int s = 0; s < STEPS; ++s) {
    const int blocks = NB * 8;                          // 8 pairs * 64 mtiles / 8 waves
    k_inc <<<blocks, 256, 0, stream>>>(seq16, SEQT, 0, s + 5, 0, WincT, biasInc, tcrow, tcT);
    k_hop <<<blocks, 256, 0, stream>>>(a1s, tcT,  tcrow, h1arow, h1aT);
    k_hop <<<blocks, 256, 0, stream>>>(a2s, tcT,  tcrow, h1brow, h1bT);
    k_hop <<<blocks, 256, 0, stream>>>(a1s, h1aT, tcrow, h2arow, (f16*)nullptr);
    k_hop <<<blocks, 256, 0, stream>>>(a2s, h1bT, tcrow, h2brow, (f16*)nullptr);
    k_proj<<<blocks, 256, 0, stream>>>(tcrow, h1arow, h2arow, h1brow, h2brow, WprojT, biasProj,
                                       1, s, 0, resF32, seq16, rec16);
  }

  // -------- decode prediction --------
  k_dec1<<<(unsigned)((PROWS / 16) * 2 / 8), 256, 0, stream>>>(seq16, decW1T, dec_b1, hidden16,
                                                               STEPS, SEQT, MAXKT);
  k_dec2<<<(unsigned)(PROWS / 16 / 8), 256, 0, stream>>>(hidden16, decW2T, dec_b2, (float*)d_out,
                                                         STEPS, 96, (unsigned long long)NB * 336 * NN);
}
